// Attention_22127671509458
// MI455X (gfx1250) — matrix operand order repaired and checked by reading, not compiled, not run on hardware
//
#include <hip/hip_runtime.h>
#include <hip/hip_bf16.h>

typedef __bf16 bf16_t;
typedef __attribute__((ext_vector_type(16))) __bf16 v16bf;
typedef __attribute__((ext_vector_type(8)))  float  v8f;
typedef __attribute__((ext_vector_type(4)))  unsigned int u32x4;
typedef __attribute__((ext_vector_type(8)))  int i32x8;
typedef __attribute__((ext_vector_type(4)))  int i32x4;

#define B_   4
#define S_   2048
#define D_   2048
#define H_   16
#define KVH_ 8
#define HD_  128
#define REP_ 2

// ---------------------------------------------------------------------------
// Tensor Data Mover: 2-D tile load (global -> LDS), bf16 elements.
// D# layout per CDNA5 ISA ch.8: group0 = {count, lds_addr, global_addr, type=2},
// group1 = {mask=0, data_size=1(2B), tensor_dim0/1, tile_dim0/1, dim0_stride}.
__device__ inline void tdm_load_2d(unsigned lds_off, const bf16_t* gptr,
                                   unsigned tensor_d0, unsigned tensor_d1,
                                   unsigned tile_d0, unsigned tile_d1,
                                   unsigned stride0_elems) {
  const unsigned long long ga = (unsigned long long)(uintptr_t)gptr;
  u32x4 g0;
  g0[0] = 1u;                                   // count=1 (valid), user mode
  g0[1] = lds_off;                              // lds_addr (byte offset)
  g0[2] = (unsigned)(ga & 0xffffffffu);         // global_addr[31:0]
  g0[3] = (unsigned)((ga >> 32) & 0x01ffffffu)  // global_addr[56:32]
        | (2u << 30);                           // type = 2 ("image")
  i32x8 g1;
  g1[0] = (int)(1u << 16);                      // workgroup_mask=0, data_size=1 (2B)
  g1[1] = (int)((tensor_d0 & 0xffffu) << 16);   // tensor_dim0[15:0]
  g1[2] = (int)(((tensor_d0 >> 16) & 0xffffu) | ((tensor_d1 & 0xffffu) << 16));
  g1[3] = (int)(((tensor_d1 >> 16) & 0xffffu) | ((tile_d0 & 0xffffu) << 16));
  g1[4] = (int)(tile_d1 & 0xffffu);             // tile_dim1, tile_dim2=0
  g1[5] = (int)stride0_elems;                   // tensor_dim0_stride[31:0]
  g1[6] = 0;
  g1[7] = 0;
  i32x4 z4 = {0, 0, 0, 0};
  i32x8 z8 = {0, 0, 0, 0, 0, 0, 0, 0};
#if __has_builtin(__builtin_amdgcn_tensor_load_to_lds)
  __builtin_amdgcn_tensor_load_to_lds(g0, g1, z4, z4, z8, 0);
#endif
}

// ---------------------------------------------------------------------------
// WMMA fragment loaders (gfx1250 wave32 layouts, 16x16x32 bf16).
// A (16x32, row-major src, stride ld): contiguous 16 bf16 per lane (k = 16h + j).
__device__ inline v16bf frag_a_load(const bf16_t* p, int ld) {
  const int lane = threadIdx.x & 31;
  const int m    = lane & 15;
  const int kb8  = (lane >> 4) << 4;   // 0 or 16
  const bf16_t* row = p + m * ld;
  v16bf a;
#pragma unroll
  for (int j = 0; j < 8; ++j) a[j]     = row[kb8 + j];
#pragma unroll
  for (int j = 0; j < 8; ++j) a[8 + j] = row[kb8 + 8 + j];
  return a;
}

// B from transposed storage: logical B(k,n) = p[n*ld + k] (contiguous in k).
__device__ inline v16bf frag_bT_load(const bf16_t* p, int ld) {
  const int lane = threadIdx.x & 31;
  const int n    = lane & 15;
  const int kb   = (lane >> 4) << 4;   // 0 or 16
  v16bf b;
#pragma unroll
  for (int j = 0; j < 16; ++j) b[j] = p[n * ld + kb + j];
  return b;
}

// ---------------------------------------------------------------------------
__global__ void f32_to_bf16_kernel(const float* __restrict__ in,
                                   bf16_t* __restrict__ out, size_t n) {
  size_t i = (size_t)blockIdx.x * blockDim.x + threadIdx.x;
  size_t stride = (size_t)gridDim.x * blockDim.x;
  for (; i < n; i += stride) out[i] = (bf16_t)in[i];
}

// W (K x N) f32 -> Wt (N x K) bf16  (coalesced read, strided write; one-time)
__global__ void transpose_cvt_kernel(const float* __restrict__ in,
                                     bf16_t* __restrict__ out, int K, int N) {
  size_t total = (size_t)K * N;
  size_t i = (size_t)blockIdx.x * blockDim.x + threadIdx.x;
  size_t stride = (size_t)gridDim.x * blockDim.x;
  for (; i < total; i += stride) {
    const int n = (int)(i % N);
    const int k = (int)(i / N);
    out[(size_t)n * K + k] = (bf16_t)in[i];
  }
}

// ---------------------------------------------------------------------------
// C = A(MxK) * Bt(NxK)^T, bf16 in, fp32 accumulate; out bf16 (Cb) or f32 (Cf).
// Block tile 128x128, wave tile 32x64, K-step 32, TDM double-buffered LDS.
// Both LDS tiles are 128x32 row-major => all fragment feeds are b128 loads.
__global__ __launch_bounds__(256)
void gemm_bf16_kernel(const bf16_t* __restrict__ A, const bf16_t* __restrict__ Bt,
                      bf16_t* __restrict__ Cb, float* __restrict__ Cf,
                      int M, int N, int K) {
  __shared__ bf16_t As[2][128 * 32];
  __shared__ bf16_t Bs[2][128 * 32];

  const int tid  = threadIdx.x;
  const int w    = tid >> 5;
  const int lane = tid & 31;
  const int wm   = (w >> 1) * 32;          // 0,32,64,96
  const int wn   = (w & 1) * 64;           // 0,64
  const int m0   = blockIdx.y * 128;
  const int n0   = blockIdx.x * 128;

  v8f acc[2][4] = {};

  const int NT = K >> 5;
  if (w == 0) {  // wave-uniform: wave 0 drives the TDM
    tdm_load_2d((unsigned)(uintptr_t)&As[0][0], A + (size_t)m0 * K,
                (unsigned)K, (unsigned)M, 32u, 128u, (unsigned)K);
    tdm_load_2d((unsigned)(uintptr_t)&Bs[0][0], Bt + (size_t)n0 * K,
                (unsigned)K, (unsigned)N, 32u, 128u, (unsigned)K);
  }

  for (int kt = 0; kt < NT; ++kt) {
    const int cur = kt & 1;
    const int nxt = cur ^ 1;
    if (w == 0) {
      if (kt + 1 < NT) {
        const int k1 = (kt + 1) << 5;
        tdm_load_2d((unsigned)(uintptr_t)&As[nxt][0], A + (size_t)m0 * K + k1,
                    (unsigned)K, (unsigned)M, 32u, 128u, (unsigned)K);
        tdm_load_2d((unsigned)(uintptr_t)&Bs[nxt][0], Bt + (size_t)n0 * K + k1,
                    (unsigned)K, (unsigned)N, 32u, 128u, (unsigned)K);
        __builtin_amdgcn_s_wait_tensorcnt(2);  // current buffer's 2 ops done (in-order)
      } else {
        __builtin_amdgcn_s_wait_tensorcnt(0);
      }
    }
    __syncthreads();

    v16bf af0 = frag_a_load(&As[cur][(wm + 0)  * 32], 32);
    v16bf af1 = frag_a_load(&As[cur][(wm + 16) * 32], 32);
#pragma unroll
    for (int j = 0; j < 4; ++j) {
      v16bf bfj = frag_bT_load(&Bs[cur][(wn + j * 16) * 32], 32);
      acc[0][j] = __builtin_amdgcn_wmma_f32_16x16x32_bf16(false, af0, false, bfj, (short)0, acc[0][j], false, false);
      acc[1][j] = __builtin_amdgcn_wmma_f32_16x16x32_bf16(false, af1, false, bfj, (short)0, acc[1][j], false, false);
    }
    __syncthreads();  // all waves done with buf[cur] before it is overwritten
  }

  const int nn = lane & 15;
  const int mb = (lane >> 4) << 3;
#pragma unroll
  for (int i = 0; i < 2; ++i) {
#pragma unroll
    for (int j = 0; j < 4; ++j) {
      const int grow = m0 + wm + i * 16 + mb;
      const int gcol = n0 + wn + j * 16 + nn;
      if (Cf) {
#pragma unroll
        for (int r = 0; r < 8; ++r) Cf[(size_t)(grow + r) * N + gcol] = acc[i][j][r];
      } else {
#pragma unroll
        for (int r = 0; r < 8; ++r) Cb[(size_t)(grow + r) * N + gcol] = (bf16_t)acc[i][j][r];
      }
    }
  }
}

// ---------------------------------------------------------------------------
// (B,S,nh,HD) -> Q/K: (B,nh,S,HD) with RoPE;  V: (B,nh,HD,S) transposed copy.
__global__ void rope_transpose_kernel(const bf16_t* __restrict__ proj,
                                      const float* __restrict__ cosb,
                                      const float* __restrict__ sinb,
                                      bf16_t* __restrict__ out,
                                      int nheads, int mode /*0=V-transpose,1=rope*/) {
  const size_t total = (size_t)B_ * S_ * nheads * (HD_ / 2);
  size_t i = (size_t)blockIdx.x * blockDim.x + threadIdx.x;
  const size_t stride = (size_t)gridDim.x * blockDim.x;
  for (; i < total; i += stride) {
    const int d2 = (int)(i % (HD_ / 2));
    size_t t = i / (HD_ / 2);
    const int hh = (int)(t % nheads); t /= nheads;
    const int s  = (int)(t % S_);
    const int b  = (int)(t / S_);
    const size_t ibase = (((size_t)(b * S_ + s) * nheads + hh) * HD_) + 2 * d2;
    float x0 = (float)proj[ibase];
    float x1 = (float)proj[ibase + 1];
    if (mode) {  // RoPE, (B,nh,S,HD) output
      const float c  = cosb[s * (HD_ / 2) + d2];
      const float sn = sinb[s * (HD_ / 2) + d2];
      const float o0 = x0 * c - x1 * sn;
      const float o1 = x0 * sn + x1 * c;
      const size_t obase = (((size_t)(b * nheads + hh) * S_ + s) * HD_) + 2 * d2;
      out[obase]     = (bf16_t)o0;
      out[obase + 1] = (bf16_t)o1;
    } else {     // V: (B,nh,HD,S) transposed output (hd-major)
      const size_t hb = ((size_t)(b * nheads + hh) * HD_);
      out[(hb + 2 * d2) * S_ + s]     = (bf16_t)x0;
      out[(hb + 2 * d2 + 1) * S_ + s] = (bf16_t)x1;
    }
  }
}

// ---------------------------------------------------------------------------
// Flash attention: block = (b,h) x 128 query rows; wave = 16 rows, private
// online softmax. K (S-major) and V^T (HD-major) tiles staged in LDS via TDM.
__global__ __launch_bounds__(256)
void flash_attn_kernel(const bf16_t* __restrict__ Q, const bf16_t* __restrict__ Kg,
                       const bf16_t* __restrict__ Vg, bf16_t* __restrict__ Og) {
  __shared__ bf16_t Kt[32 * 128];   // [key][hd]
  __shared__ bf16_t Vt[128 * 32];   // [hd][key]
  __shared__ bf16_t Ps[8][16 * 32];

  const int tid  = threadIdx.x;
  const int w    = tid >> 5;
  const int lane = tid & 31;
  const int bh   = blockIdx.y;
  const int b    = bh / H_;
  const int h    = bh % H_;
  const int kvh  = h / REP_;
  const int q0b  = blockIdx.x * 128;
  const int q0   = q0b + w * 16;

  const bf16_t* Qg = Q + (((size_t)(b * H_ + h) * S_) + q0) * HD_;
  v16bf qf[4];
#pragma unroll
  for (int c = 0; c < 4; ++c) qf[c] = frag_a_load(Qg + c * 32, HD_);

  v8f o[8] = {};
  float mi[8], li[8];
#pragma unroll
  for (int r = 0; r < 8; ++r) { mi[r] = -1e30f; li[r] = 0.f; }

  const int nn = lane & 15;
  const int mb = (lane >> 4) << 3;
  const int rowg = q0 + mb;
  const int wave_row_max = q0 + 15;

  const bf16_t* Kbase = Kg + ((size_t)(b * KVH_ + kvh) * S_) * HD_;   // [s][d]
  const bf16_t* Vbase = Vg + ((size_t)(b * KVH_ + kvh) * HD_) * S_;   // [d][s]
  const float scale = 0.08838834764831845f;  // 1/sqrt(128)

  const unsigned kt_off = (unsigned)(uintptr_t)&Kt[0];
  const unsigned vt_off = (unsigned)(uintptr_t)&Vt[0];

  for (int kv0 = 0; kv0 < q0b + 128; kv0 += 32) {
    if (w == 0) {
      // K: 32 rows (s) x 128 (d), row-major
      tdm_load_2d(kt_off, Kbase + (size_t)kv0 * HD_, (unsigned)HD_, (unsigned)S_,
                  (unsigned)HD_, 32u, (unsigned)HD_);
      // V^T: 128 rows (d) x 32 (s), stride S
      tdm_load_2d(vt_off, Vbase + kv0, (unsigned)S_, (unsigned)HD_,
                  32u, 128u, (unsigned)S_);
      __builtin_amdgcn_s_wait_tensorcnt(0);
    }
    __syncthreads();

    if (kv0 <= wave_row_max) {  // wave-uniform branch: EXEC all-ones inside
      v8f s0 = {}, s1 = {};
#pragma unroll
      for (int c = 0; c < 4; ++c) {
        v16bf kb0 = frag_bT_load(Kt + 0 * 128 + c * 32, 128);   // keys 0..15
        v16bf kb1 = frag_bT_load(Kt + 16 * 128 + c * 32, 128);  // keys 16..31
        s0 = __builtin_amdgcn_wmma_f32_16x16x32_bf16(false, qf[c], false, kb0, (short)0, s0, false, false);
        s1 = __builtin_amdgcn_wmma_f32_16x16x32_bf16(false, qf[c], false, kb1, (short)0, s1, false, false);
      }

      float mx[8];
#pragma unroll
      for (int r = 0; r < 8; ++r) {
        const int row = rowg + r;
        float v0 = s0[r] * scale + (((kv0 + nn)      > row) ? -1e30f : 0.f);
        float v1 = s1[r] * scale + (((kv0 + 16 + nn) > row) ? -1e30f : 0.f);
        s0[r] = v0; s1[r] = v1;
        mx[r] = fmaxf(v0, v1);
      }
#pragma unroll
      for (int off = 1; off < 16; off <<= 1)
#pragma unroll
        for (int r = 0; r < 8; ++r) mx[r] = fmaxf(mx[r], __shfl_xor(mx[r], off, 32));

      float rs[8];
#pragma unroll
      for (int r = 0; r < 8; ++r) {
        const float mnew  = fmaxf(mi[r], mx[r]);
        const float alpha = __expf(mi[r] - mnew);
        mi[r] = mnew;
        s0[r] = __expf(s0[r] - mnew);
        s1[r] = __expf(s1[r] - mnew);
        rs[r] = s0[r] + s1[r];
        li[r] *= alpha;
#pragma unroll
        for (int c = 0; c < 8; ++c) o[c][r] *= alpha;
      }
#pragma unroll
      for (int off = 1; off < 16; off <<= 1)
#pragma unroll
        for (int r = 0; r < 8; ++r) rs[r] += __shfl_xor(rs[r], off, 32);
#pragma unroll
      for (int r = 0; r < 8; ++r) li[r] += rs[r];

      // P: C-layout f32 -> per-wave LDS scratch -> A-layout bf16
      bf16_t* ps = Ps[w];
#pragma unroll
      for (int r = 0; r < 8; ++r) {
        ps[(mb + r) * 32 + nn]      = (bf16_t)s0[r];
        ps[(mb + r) * 32 + 16 + nn] = (bf16_t)s1[r];
      }
      v16bf pf = frag_a_load(ps, 32);
#pragma unroll
      for (int c = 0; c < 8; ++c) {
        v16bf vf = frag_bT_load(Vt + (c * 16) * 32, 32);  // B(k=key, n=hd)
        o[c] = __builtin_amdgcn_wmma_f32_16x16x32_bf16(false, pf, false, vf, (short)0, o[c], false, false);
      }
    }
    __syncthreads();
  }

  // epilogue: (B,S,H*HD) bf16
#pragma unroll
  for (int c = 0; c < 8; ++c) {
#pragma unroll
    for (int r = 0; r < 8; ++r) {
      Og[((size_t)(b * S_) + q0 + mb + r) * (H_ * HD_) + h * HD_ + c * 16 + nn] =
          (bf16_t)(o[c][r] / li[r]);
    }
  }
}

// ---------------------------------------------------------------------------
extern "C" void kernel_launch(void* const* d_in, const int* in_sizes, int n_in,
                              void* d_out, int out_size, void* d_ws, size_t ws_size,
                              hipStream_t stream) {
  (void)in_sizes; (void)n_in; (void)out_size; (void)ws_size;
  const float* x  = (const float*)d_in[0];
  const float* fc = (const float*)d_in[1];
  const float* fs = (const float*)d_in[2];
  const float* Wq = (const float*)d_in[3];
  const float* Wk = (const float*)d_in[4];
  const float* Wv = (const float*)d_in[5];
  const float* Wo = (const float*)d_in[6];
  float* out = (float*)d_out;

  bf16_t* wsb = (bf16_t*)d_ws;
  size_t off = 0;
  bf16_t* xb    = wsb + off; off += (size_t)B_ * S_ * D_;          // reused as Qr
  bf16_t* wqb   = wsb + off; off += (size_t)D_ * H_ * HD_;         // Wq^T (N x K)
  bf16_t* wkb   = wsb + off; off += (size_t)D_ * KVH_ * HD_;       // Wk^T
  bf16_t* wvb   = wsb + off; off += (size_t)D_ * KVH_ * HD_;       // Wv^T
  bf16_t* wob   = wsb + off; off += (size_t)H_ * HD_ * D_;         // Wo^T
  bf16_t* qproj = wsb + off; off += (size_t)B_ * S_ * H_ * HD_;    // reused as attn
  bf16_t* kproj = wsb + off; off += (size_t)B_ * S_ * KVH_ * HD_;
  bf16_t* vproj = wsb + off; off += (size_t)B_ * S_ * KVH_ * HD_;
  bf16_t* Kr    = wsb + off; off += (size_t)B_ * KVH_ * S_ * HD_;
  bf16_t* Vr    = wsb + off; off += (size_t)B_ * KVH_ * HD_ * S_;  // V^T (hd-major)
  bf16_t* Qr    = xb;     // x no longer needed after projections
  bf16_t* attn  = qproj;  // qproj no longer needed after rope

  {  // x -> bf16 (no transpose)
    size_t n = (size_t)B_ * S_ * D_;
    f32_to_bf16_kernel<<<dim3((int)((n + 255) / 256)), dim3(256), 0, stream>>>(x, xb, n);
  }
  auto tcvt = [&](const float* src, bf16_t* dst, int K, int N) {
    size_t n = (size_t)K * N;
    transpose_cvt_kernel<<<dim3((int)((n + 255) / 256)), dim3(256), 0, stream>>>(src, dst, K, N);
  };
  tcvt(Wq, wqb, D_, H_ * HD_);
  tcvt(Wk, wkb, D_, KVH_ * HD_);
  tcvt(Wv, wvb, D_, KVH_ * HD_);
  tcvt(Wo, wob, H_ * HD_, D_);

  const int M = B_ * S_;  // 8192

  gemm_bf16_kernel<<<dim3((H_ * HD_) / 128, M / 128), 256, 0, stream>>>(
      xb, wqb, qproj, nullptr, M, H_ * HD_, D_);
  gemm_bf16_kernel<<<dim3((KVH_ * HD_) / 128, M / 128), 256, 0, stream>>>(
      xb, wkb, kproj, nullptr, M, KVH_ * HD_, D_);
  gemm_bf16_kernel<<<dim3((KVH_ * HD_) / 128, M / 128), 256, 0, stream>>>(
      xb, wvb, vproj, nullptr, M, KVH_ * HD_, D_);

  {
    size_t qp = (size_t)B_ * S_ * H_ * (HD_ / 2);
    size_t kp = (size_t)B_ * S_ * KVH_ * (HD_ / 2);
    rope_transpose_kernel<<<dim3((int)((qp + 255) / 256)), dim3(256), 0, stream>>>(
        qproj, fc, fs, Qr, H_, 1);
    rope_transpose_kernel<<<dim3((int)((kp + 255) / 256)), dim3(256), 0, stream>>>(
        kproj, fc, fs, Kr, KVH_, 1);
    rope_transpose_kernel<<<dim3((int)((kp + 255) / 256)), dim3(256), 0, stream>>>(
        vproj, fc, fs, Vr, KVH_, 0);
  }

  flash_attn_kernel<<<dim3(S_ / 128, B_ * H_), 256, 0, stream>>>(Qr, Kr, Vr, attn);

  gemm_bf16_kernel<<<dim3(D_ / 128, M / 128), 256, 0, stream>>>(
      attn, wob, nullptr, out, M, D_, H_ * HD_);
}
